// BailingMoELinearDecoderLayer_721554506406
// MI455X (gfx1250) — matrix operand order repaired and checked by reading, not compiled, not run on hardware
//
#include <hip/hip_runtime.h>
#include <hip/hip_bf16.h>
#include <math.h>

// ---------------- problem constants ----------------
#define T_TOK 1024
#define HDIM  2048
#define NHQ   16
#define NKVH  4
#define HDH   128
#define NE    32
#define TOPK  4
#define IDIM  1024
#define QKVD  3072              // (NHQ + 2*NKVH)*HDH
#define EPSN  1e-6f
#define THETAF 600000.0f
#define ATTN_SCALE 0.08838834764831845f   // 128^-0.5

typedef __attribute__((ext_vector_type(16))) __bf16 v16bf;
typedef __attribute__((ext_vector_type(8)))  __bf16 v8bf;
typedef __attribute__((ext_vector_type(8)))  float  v8f;

// LDS byte-offset of a generic pointer known to live in LDS (AS3 ptrs are
// 32-bit offsets from the workgroup LDS base on AMDGPU).
__device__ __forceinline__ unsigned to_lds(const void* p) {
  return (unsigned)(unsigned long long)(const __attribute__((address_space(3))) char*)p;
}

// ---------------- generic bf16 WMMA GEMM ----------------
// C = alpha * A(bf16,[M,K]) * B(f32 or bf16, element(n,k) at n*sN+k*sK) [+ residual]
// Tile 128x64x64, 256 threads = 8 waves, each wave a 32x32 patch:
// 2x2 of v_wmma_f32_16x16x32_bf16, two K sub-steps -> 8 WMMAs per barrier.
// Double-buffered LDS; A staged via GLOBAL_LOAD_ASYNC_TO_LDS_B128 (ASYNCcnt)
// in the dense path, register-staged in the MoE path (ragged M / row gather);
// B register-staged with on-the-fly fp32->bf16 conversion.
#define BM 128
#define BN 64
#define BK 64
#define LDSP 72   // padded LDS row (elements); 144B row stride keeps 16B alignment

struct GemmP {
  const __bf16* A; long lda; long sAz;
  const void*   B; long sN;  long sK; long sBz; int zdivB;
  float*        C; long ldc; long sCz;
  long M, N, K;
  float alpha;
  const float* R; long ldr;
  const int* cnt; const int* off; const int* gather;  // MoE mode
};

template<bool B_BF16, bool MOE, bool GATHER, bool RES>
__global__ __launch_bounds__(256) void gemm_kernel(GemmP p) {
  __shared__ __bf16 lds_a[2][BM * LDSP];
  __shared__ __bf16 lds_b[2][BN * LDSP];

  const int tid = threadIdx.x;
  const int z   = blockIdx.z;

  long Mloc = p.M;
  long rowbase = 0;
  const __bf16* A  = p.A;
  const float*  Bf = (const float*)p.B;
  const __bf16* Bh = (const __bf16*)p.B;
  float* C = p.C;

  if constexpr (MOE) {
    Mloc    = (long)p.cnt[z];
    rowbase = (long)p.off[z];
    Bf += (long)z * p.sBz;                 // per-expert fp32 weights
  } else {
    A  += (long)z * p.sAz;
    long bo = (long)(z / p.zdivB) * p.sBz; // GQA: kv head = q head / 4
    Bf += bo; Bh += bo;
    C  += (long)z * p.sCz;
  }

  const long m0 = (long)blockIdx.y * BM;
  if (m0 >= Mloc) return;                  // uniform per block (MoE tail tiles)
  const long n0 = (long)blockIdx.x * BN;

  const int wid = tid >> 5, lane = tid & 31;
  const int wm = wid >> 1, wn = wid & 1;   // 4x2 wave grid
  const int g  = lane >> 4, l16 = lane & 15;

  // staging registers
  v8bf   aR[4];
  float  bRf[16];
  __bf16 bRh[16];

  // ---- stage A for buffer `b` at depth kk: async (dense) or regs (MoE) ----
  auto stageA = [&](long kk, int b) {
#pragma unroll
    for (int pp = 0; pp < 4; ++pp) {
      int chunk = pp * 256 + tid;
      int r = chunk >> 3, c = (chunk & 7) * 8;   // row 0..127, col 0,8,..,56
      if constexpr (!MOE) {
        const __bf16* ga = A + (long)(m0 + r) * p.lda + kk + c;
        unsigned la = to_lds(&lds_a[b][r * LDSP + c]);
        asm volatile("global_load_async_to_lds_b128 %0, %1, off"
                     :: "v"(la), "v"(ga) : "memory");
      } else {
        v8bf av;
#pragma unroll
        for (int q = 0; q < 8; ++q) av[q] = (__bf16)0.0f;
        if (m0 + r < Mloc) {
          long arow = GATHER ? (long)p.gather[rowbase + m0 + r]
                             : (rowbase + m0 + r);
          av = *(const v8bf*)(A + arow * p.lda + kk + c);
        }
        aR[pp] = av;
      }
    }
  };
  auto storeA_moe = [&](int b) {
    if constexpr (MOE) {
#pragma unroll
      for (int pp = 0; pp < 4; ++pp) {
        int chunk = pp * 256 + tid;
        int r = chunk >> 3, c = (chunk & 7) * 8;
        *(v8bf*)(&lds_a[b][r * LDSP + c]) = aR[pp];
      }
    }
  };
  // ---- stage B tile (fp32 or bf16 source) into registers at depth kk ----
  auto loadB = [&](long kk) {
#pragma unroll
    for (int i = 0; i < 16; ++i) {
      int e = tid * 16 + i;
      int n = e >> 6, k = e & 63;
      long idx = (n0 + n) * p.sN + (kk + k) * p.sK;
      if constexpr (B_BF16) bRh[i] = Bh[idx]; else bRf[i] = Bf[idx];
      if (i == 0 && kk + BK < p.K) {       // emits global_prefetch_b8
        if constexpr (B_BF16) __builtin_prefetch(&Bh[idx + (long)BK * p.sK], 0, 1);
        else                  __builtin_prefetch(&Bf[idx + (long)BK * p.sK], 0, 1);
      }
    }
  };
  auto storeB = [&](int b) {
#pragma unroll
    for (int i = 0; i < 16; ++i) {
      int e = tid * 16 + i;
      int n = e >> 6, k = e & 63;
      __bf16 v;
      if constexpr (B_BF16) v = bRh[i]; else v = (__bf16)bRf[i];
      lds_b[b][n * LDSP + k] = v;
    }
  };

  v8f acc[2][2];
#pragma unroll
  for (int i = 0; i < 2; ++i)
#pragma unroll
    for (int j = 0; j < 2; ++j)
#pragma unroll
      for (int q = 0; q < 8; ++q) acc[i][j][q] = 0.0f;

  // ---- software pipeline: prologue stages tile 0 ----
  stageA(0, 0);
  loadB(0);

  int buf = 0;
  for (long kk = 0; kk < p.K; kk += BK) {
    storeB(buf);
    storeA_moe(buf);
    if constexpr (!MOE)
      asm volatile("s_wait_asynccnt 0x0" ::: "memory");  // A tile landed in LDS
    __syncthreads();

    if (kk + BK < p.K) {                   // kick next tile into the other buffer
      stageA(kk + BK, buf ^ 1);
      loadB(kk + BK);
    }

    // ---- 8 WMMAs on buffer `buf` (fragments per ISA 7.12.2 layouts) ----
#pragma unroll
    for (int ks = 0; ks < BK; ks += 32) {
      v16bf af[2], bfr[2];
#pragma unroll
      for (int s2 = 0; s2 < 2; ++s2) {
        const __bf16* pa = &lds_a[buf][(wm * 32 + s2 * 16 + l16) * LDSP + ks + g * 8];
        v8bf alo = *(const v8bf*)pa;        // K = ks + g*8 ..
        v8bf ahi = *(const v8bf*)(pa + 16); // K = ks + 16+g*8 ..
#pragma unroll
        for (int q = 0; q < 8; ++q) { af[s2][q] = alo[q]; af[s2][q + 8] = ahi[q]; }
        const __bf16* pb = &lds_b[buf][(wn * 32 + s2 * 16 + l16) * LDSP + ks + g * 8];
        v8bf blo = *(const v8bf*)pb;        // K = ks + g*8 ..
        v8bf bhi = *(const v8bf*)(pb + 16);
#pragma unroll
        for (int q = 0; q < 8; ++q) { bfr[s2][q] = blo[q]; bfr[s2][q + 8] = bhi[q]; }
      }
#pragma unroll
      for (int sm = 0; sm < 2; ++sm)
#pragma unroll
        for (int sn = 0; sn < 2; ++sn)
          acc[sm][sn] = __builtin_amdgcn_wmma_f32_16x16x32_bf16(
              false, af[sm], false, bfr[sn], (short)0, acc[sm][sn], false, false);
    }
    buf ^= 1;   // double-buffered: no trailing barrier needed
  }

  // ---- epilogue: C layout lane=N, vgpr r -> M = r + 8*g ----
#pragma unroll
  for (int sm = 0; sm < 2; ++sm) {
    long mbase = m0 + wm * 32 + sm * 16 + g * 8;
#pragma unroll
    for (int sn = 0; sn < 2; ++sn) {
      long ncol = n0 + wn * 32 + sn * 16 + l16;
#pragma unroll
      for (int r2 = 0; r2 < 8; ++r2) {
        long mr = mbase + r2;
        if (mr < Mloc) {
          float v = p.alpha * acc[sm][sn][r2];
          if constexpr (RES) v += p.R[mr * p.ldr + ncol];
          C[(rowbase + mr) * p.ldc + ncol] = v;
        }
      }
    }
  }
}

// ---------------- RMSNorm (per-token block) ----------------
__global__ __launch_bounds__(256) void rmsnorm_kernel(const float* __restrict__ x,
                                                      const float* __restrict__ w,
                                                      __bf16* __restrict__ ob,
                                                      float* __restrict__ of) {
  __shared__ float red[256];
  long t = blockIdx.x;
  const float* row = x + t * HDIM;
  float ss = 0.f;
  for (int i = threadIdx.x; i < HDIM; i += 256) { float v = row[i]; ss += v * v; }
  red[threadIdx.x] = ss; __syncthreads();
  for (int s = 128; s > 0; s >>= 1) {
    if (threadIdx.x < s) red[threadIdx.x] += red[threadIdx.x + s];
    __syncthreads();
  }
  float scale = rsqrtf(red[0] / (float)HDIM + EPSN);
  for (int i = threadIdx.x; i < HDIM; i += 256) {
    float v = row[i] * scale * w[i];
    ob[t * HDIM + i] = (__bf16)v;
    if (of) of[t * HDIM + i] = v;
  }
}

// ---------------- RoPE on q,k + bf16 conversion of q,k,v ----------------
__global__ __launch_bounds__(256) void rope_kernel(const float* __restrict__ qkv,
                                                   const int* __restrict__ pos,
                                                   __bf16* __restrict__ qb,
                                                   __bf16* __restrict__ kb,
                                                   __bf16* __restrict__ vb) {
  long t = blockIdx.x;
  const float* row = qkv + t * QKVD;
  float pf = (float)pos[t];
  for (int idx = threadIdx.x; idx < (NHQ + NKVH) * 64; idx += 256) {
    int head = idx / 64;            // 0..15 q, 16..19 k (contiguous in qkv)
    int d2 = idx % 64;
    float invf = powf(THETAF, -(float)d2 / 64.0f);
    float ang = pf * invf;
    float c = cosf(ang), s = sinf(ang);
    const float* hp = row + head * HDH;
    float x1 = hp[d2], x2 = hp[d2 + 64];
    float o1 = x1 * c - x2 * s;
    float o2 = x1 * s + x2 * c;
    if (head < NHQ) {
      __bf16* q = qb + t * (NHQ * HDH) + head * HDH;
      q[d2] = (__bf16)o1; q[d2 + 64] = (__bf16)o2;
    } else {
      __bf16* kx = kb + t * (NKVH * HDH) + (head - NHQ) * HDH;
      kx[d2] = (__bf16)o1; kx[d2 + 64] = (__bf16)o2;
    }
  }
  for (int idx = threadIdx.x; idx < NKVH * HDH; idx += 256)
    vb[t * (NKVH * HDH) + idx] = (__bf16)row[NHQ * HDH + NKVH * HDH + idx];
}

// ---------------- causal softmax: f32 scores -> bf16 P ----------------
__global__ __launch_bounds__(256) void softmax_kernel(const float* __restrict__ S,
                                                      __bf16* __restrict__ P) {
  __shared__ float red[256];
  long t = blockIdx.x, h = blockIdx.y;
  const float* srow = S + (h * T_TOK + t) * (long)T_TOK;
  __bf16* prow = P + (h * T_TOK + t) * (long)T_TOK;
  int len = (int)t + 1;
  float m = -3.4e38f;
  for (int i = threadIdx.x; i < len; i += 256) m = fmaxf(m, srow[i]);
  red[threadIdx.x] = m; __syncthreads();
  for (int s = 128; s > 0; s >>= 1) {
    if (threadIdx.x < s) red[threadIdx.x] = fmaxf(red[threadIdx.x], red[threadIdx.x + s]);
    __syncthreads();
  }
  m = red[0]; __syncthreads();
  float sum = 0.f;
  for (int i = threadIdx.x; i < len; i += 256) sum += __expf(srow[i] - m);
  red[threadIdx.x] = sum; __syncthreads();
  for (int s = 128; s > 0; s >>= 1) {
    if (threadIdx.x < s) red[threadIdx.x] += red[threadIdx.x + s];
    __syncthreads();
  }
  float inv = 1.0f / red[0];
  for (int i = threadIdx.x; i < T_TOK; i += 256)
    prow[i] = (i < len) ? (__bf16)(__expf(srow[i] - m) * inv) : (__bf16)0.0f;
}

// ---------------- f32 -> bf16 conversion ----------------
__global__ __launch_bounds__(256) void cvt_kernel(const float* __restrict__ in,
                                                  __bf16* __restrict__ out, long n) {
  long i = (long)blockIdx.x * 256 + threadIdx.x;
  if (i < n) out[i] = (__bf16)in[i];
}

// ---------------- router: logits, softmax, top-4, normalize ----------------
__global__ __launch_bounds__(256) void router_kernel(const float* __restrict__ h2,
                                                     const float* __restrict__ wr,
                                                     int* __restrict__ topi,
                                                     float* __restrict__ topw) {
  __shared__ float sh[HDIM];
  __shared__ float red[256];
  __shared__ float logits[NE];
  long t = blockIdx.x;
  for (int i = threadIdx.x; i < HDIM; i += 256) sh[i] = h2[t * HDIM + i];
  __syncthreads();
  int e = threadIdx.x >> 3, seg = threadIdx.x & 7;  // 32 experts x 8 segments
  const float* w = wr + (long)e * HDIM + seg * 256;
  float s = 0.f;
  for (int i = 0; i < 256; ++i) s += sh[seg * 256 + i] * w[i];
  red[threadIdx.x] = s; __syncthreads();
  if (seg == 0) {
    float l = 0.f;
    for (int j = 0; j < 8; ++j) l += red[e * 8 + j];
    logits[e] = l;
  }
  __syncthreads();
  if (threadIdx.x == 0) {
    float m = logits[0];
    for (int i = 1; i < NE; ++i) m = fmaxf(m, logits[i]);
    float Z = 0.f, p[NE];
    for (int i = 0; i < NE; ++i) { p[i] = expf(logits[i] - m); Z += p[i]; }
    for (int i = 0; i < NE; ++i) p[i] /= Z;
    int used[NE]; for (int i = 0; i < NE; ++i) used[i] = 0;
    float tw[TOPK]; float wsum = 0.f;
    for (int slot = 0; slot < TOPK; ++slot) {
      int best = -1; float bv = -1.f;
      for (int i = 0; i < NE; ++i)
        if (!used[i] && p[i] > bv) { bv = p[i]; best = i; }
      used[best] = 1;
      topi[t * TOPK + slot] = best;
      tw[slot] = bv; wsum += bv;
    }
    for (int slot = 0; slot < TOPK; ++slot)
      topw[t * TOPK + slot] = tw[slot] / wsum;
  }
}

// ---------------- routing bookkeeping ----------------
__global__ void zero_kernel(int* p, int n) {
  int i = blockIdx.x * blockDim.x + threadIdx.x;
  if (i < n) p[i] = 0;
}
__global__ void count_kernel(const int* __restrict__ topi, int* __restrict__ cnt) {
  int id = blockIdx.x * blockDim.x + threadIdx.x;
  if (id < T_TOK * TOPK) atomicAdd(&cnt[topi[id]], 1);
}
__global__ void scan_kernel(const int* __restrict__ cnt, int* __restrict__ off,
                            int* __restrict__ cur) {
  if (threadIdx.x == 0) {
    int acc = 0;
    for (int e = 0; e < NE; ++e) { off[e] = acc; cur[e] = acc; acc += cnt[e]; }
  }
}
__global__ void place_kernel(const int* __restrict__ topi, int* __restrict__ cur,
                             int* __restrict__ gtok, int* __restrict__ rowof) {
  int id = blockIdx.x * blockDim.x + threadIdx.x;
  if (id < T_TOK * TOPK) {
    int e = topi[id];
    int pos = atomicAdd(&cur[e], 1);
    gtok[pos] = id >> 2;      // token index
    rowof[id] = pos;          // inverse map -> deterministic combine
  }
}

// ---------------- activations ----------------
__device__ __forceinline__ float silu(float x) { return x / (1.0f + __expf(-x)); }

__global__ __launch_bounds__(256) void moe_act_kernel(const float* __restrict__ g,
                                                      const float* __restrict__ u,
                                                      __bf16* __restrict__ a) {
  long i = (long)blockIdx.x * 256 + threadIdx.x;
  if (i < (long)T_TOK * TOPK * IDIM) a[i] = (__bf16)(silu(g[i]) * u[i]);
}
__global__ __launch_bounds__(256) void shared_act_kernel(const float* __restrict__ gu,
                                                         __bf16* __restrict__ a) {
  long i = (long)blockIdx.x * 256 + threadIdx.x;
  if (i < (long)T_TOK * IDIM) {
    long t = i / IDIM, c = i % IDIM;
    float gs = gu[t * (2 * IDIM) + c];
    float us = gu[t * (2 * IDIM) + IDIM + c];
    a[i] = (__bf16)(silu(gs) * us);
  }
}

// ---------------- final combine: out += sum_slot w * d[row_of] ----------------
__global__ __launch_bounds__(256) void combine_kernel(float* __restrict__ out,
                                                      const float* __restrict__ d,
                                                      const int* __restrict__ rowof,
                                                      const float* __restrict__ topw) {
  long i = (long)blockIdx.x * 256 + threadIdx.x;
  if (i < (long)T_TOK * HDIM) {
    long t = i / HDIM, c = i % HDIM;
    float v = out[i];
#pragma unroll
    for (int slot = 0; slot < TOPK; ++slot)
      v += topw[t * TOPK + slot] * d[(long)rowof[t * TOPK + slot] * HDIM + c];
    out[i] = v;
  }
}

// ---------------- host-side dispatch ----------------
static void launch_gemm(const GemmP& p, int gx, int gy, int gz, hipStream_t s,
                        bool bbf, bool moe, bool gather, bool res) {
  dim3 g(gx, gy, gz), b(256);
  if (moe) {
    if (gather) gemm_kernel<false, true, true,  false><<<g, b, 0, s>>>(p);
    else        gemm_kernel<false, true, false, false><<<g, b, 0, s>>>(p);
  } else if (bbf) {
    if (res) gemm_kernel<true, false, false, true ><<<g, b, 0, s>>>(p);
    else     gemm_kernel<true, false, false, false><<<g, b, 0, s>>>(p);
  } else {
    if (res) gemm_kernel<false, false, false, true ><<<g, b, 0, s>>>(p);
    else     gemm_kernel<false, false, false, false><<<g, b, 0, s>>>(p);
  }
}

extern "C" void kernel_launch(void* const* d_in, const int* in_sizes, int n_in,
                              void* d_out, int out_size, void* d_ws, size_t ws_size,
                              hipStream_t stream) {
  const float* x       = (const float*)d_in[0];
  const int*   pos     = (const int*)d_in[1];
  const float* w_rms1  = (const float*)d_in[2];
  const float* w_rms2  = (const float*)d_in[3];
  const float* w_qkv   = (const float*)d_in[4];
  const float* w_o     = (const float*)d_in[5];
  const float* w_rtr   = (const float*)d_in[6];
  const float* w1      = (const float*)d_in[7];
  const float* w3      = (const float*)d_in[8];
  const float* w2      = (const float*)d_in[9];
  const float* ws_gu   = (const float*)d_in[10];
  const float* ws_dn   = (const float*)d_in[11];
  float* out = (float*)d_out;

  // workspace carve-out
  char* wp = (char*)d_ws;
  auto alloc = [&](size_t bytes) -> void* {
    void* p = (void*)wp;
    wp += (bytes + 255) & ~(size_t)255;
    return p;
  };
  __bf16* h_bf   = (__bf16*)alloc((size_t)T_TOK * HDIM * 2);
  float*  qkv    = (float*) alloc((size_t)T_TOK * QKVD * 4);
  __bf16* qb     = (__bf16*)alloc((size_t)T_TOK * NHQ * HDH * 2);
  __bf16* kb     = (__bf16*)alloc((size_t)T_TOK * NKVH * HDH * 2);
  __bf16* vb     = (__bf16*)alloc((size_t)T_TOK * NKVH * HDH * 2);
  float*  scores = (float*) alloc((size_t)NHQ * T_TOK * T_TOK * 4);
  __bf16* Pm     = (__bf16*)alloc((size_t)NHQ * T_TOK * T_TOK * 2);
  float*  attn   = (float*) alloc((size_t)T_TOK * HDIM * 4);
  __bf16* ob     = (__bf16*)alloc((size_t)T_TOK * HDIM * 2);
  float*  x1     = (float*) alloc((size_t)T_TOK * HDIM * 4);
  float*  h2     = (float*) alloc((size_t)T_TOK * HDIM * 4);
  __bf16* h2b    = (__bf16*)alloc((size_t)T_TOK * HDIM * 2);
  int*    topi   = (int*)   alloc((size_t)T_TOK * TOPK * 4);
  float*  topw   = (float*) alloc((size_t)T_TOK * TOPK * 4);
  int*    cnt    = (int*)   alloc(NE * 4);
  int*    offs   = (int*)   alloc(NE * 4);
  int*    cur    = (int*)   alloc(NE * 4);
  int*    gtok   = (int*)   alloc((size_t)T_TOK * TOPK * 4);
  int*    rowof  = (int*)   alloc((size_t)T_TOK * TOPK * 4);
  float*  gbuf   = (float*) alloc((size_t)T_TOK * TOPK * IDIM * 4);
  float*  ubuf   = (float*) alloc((size_t)T_TOK * TOPK * IDIM * 4);
  __bf16* abuf   = (__bf16*)alloc((size_t)T_TOK * TOPK * IDIM * 2);
  float*  dmoe   = (float*) alloc((size_t)T_TOK * TOPK * HDIM * 4);
  float*  gu     = (float*) alloc((size_t)T_TOK * 2 * IDIM * 4);
  __bf16* sb     = (__bf16*)alloc((size_t)T_TOK * IDIM * 2);
  (void)ws_size; (void)in_sizes; (void)n_in; (void)out_size;

  // 1) RMSNorm1 -> h (bf16)
  rmsnorm_kernel<<<T_TOK, 256, 0, stream>>>(x, w_rms1, h_bf, nullptr);

  // 2) qkv = h @ w_qkv^T   [1024x3072], K=2048
  {
    GemmP p{h_bf, HDIM, 0, w_qkv, HDIM, 1, 0, 1, qkv, QKVD, 0,
            T_TOK, QKVD, HDIM, 1.0f, nullptr, 0, nullptr, nullptr, nullptr};
    launch_gemm(p, QKVD / BN, T_TOK / BM, 1, stream, false, false, false, false);
  }

  // 3) RoPE + bf16 q/k/v
  rope_kernel<<<T_TOK, 256, 0, stream>>>(qkv, pos, qb, kb, vb);

  // 4) scores[h] = SCALE * Q_h K_h^T   (per head, K=128)
  {
    GemmP p{qb, NHQ * HDH, HDH, kb, NKVH * HDH, 1, HDH, NHQ / NKVH,
            scores, T_TOK, (long)T_TOK * T_TOK,
            T_TOK, T_TOK, HDH, ATTN_SCALE, nullptr, 0, nullptr, nullptr, nullptr};
    launch_gemm(p, T_TOK / BN, T_TOK / BM, NHQ, stream, true, false, false, false);
  }

  // 5) causal softmax -> P (bf16)
  softmax_kernel<<<dim3(T_TOK, NHQ), 256, 0, stream>>>(scores, Pm);

  // 6) attn[h] = P_h @ V_h   (B element (n=hd,k=s) = v[s,kv,hd])
  {
    GemmP p{Pm, T_TOK, (long)T_TOK * T_TOK, vb, 1, NKVH * HDH, HDH, NHQ / NKVH,
            attn, HDIM, HDH,
            T_TOK, HDH, T_TOK, 1.0f, nullptr, 0, nullptr, nullptr, nullptr};
    launch_gemm(p, HDH / BN, T_TOK / BM, NHQ, stream, true, false, false, false);
  }

  // 7) attn -> bf16
  cvt_kernel<<<(T_TOK * HDIM) / 256, 256, 0, stream>>>(attn, ob, (long)T_TOK * HDIM);

  // 8) x1 = x + attn @ w_o^T
  {
    GemmP p{ob, HDIM, 0, w_o, HDIM, 1, 0, 1, x1, HDIM, 0,
            T_TOK, HDIM, HDIM, 1.0f, x, HDIM, nullptr, nullptr, nullptr};
    launch_gemm(p, HDIM / BN, T_TOK / BM, 1, stream, false, false, false, true);
  }

  // 9) RMSNorm2 -> h2 (f32 + bf16)
  rmsnorm_kernel<<<T_TOK, 256, 0, stream>>>(x1, w_rms2, h2b, h2);

  // 10) router + top-4
  router_kernel<<<T_TOK, 256, 0, stream>>>(h2, w_rtr, topi, topw);

  // 11-14) expert grouping (integer atomics; result deterministic via rowof)
  zero_kernel<<<1, 64, 0, stream>>>(cnt, NE);
  count_kernel<<<(T_TOK * TOPK) / 256, 256, 0, stream>>>(topi, cnt);
  scan_kernel<<<1, 32, 0, stream>>>(cnt, offs, cur);
  place_kernel<<<(T_TOK * TOPK) / 256, 256, 0, stream>>>(topi, cur, gtok, rowof);

  // 15-16) g = h2[gather] @ w1[e], u = h2[gather] @ w3[e]  (K=2048, N=1024)
  {
    GemmP p{h2b, HDIM, 0, w1, 1, IDIM, (long)HDIM * IDIM, 1, gbuf, IDIM, 0,
            T_TOK, IDIM, HDIM, 1.0f, nullptr, 0, cnt, offs, gtok};
    launch_gemm(p, IDIM / BN, T_TOK / BM, NE, stream, false, true, true, false);
    p.B = w3; p.C = ubuf;
    launch_gemm(p, IDIM / BN, T_TOK / BM, NE, stream, false, true, true, false);
  }

  // 17) a = silu(g)*u  (bf16)
  moe_act_kernel<<<((long)T_TOK * TOPK * IDIM) / 256, 256, 0, stream>>>(gbuf, ubuf, abuf);

  // 18) d = a @ w2[e]  (K=1024, N=2048), rows already expert-compacted
  {
    GemmP p{abuf, IDIM, 0, w2, 1, HDIM, (long)IDIM * HDIM, 1, dmoe, HDIM, 0,
            T_TOK, HDIM, IDIM, 1.0f, nullptr, 0, cnt, offs, nullptr};
    launch_gemm(p, HDIM / BN, T_TOK / BM, NE, stream, false, true, false, false);
  }

  // 19) shared gate_up = h2 @ ws_gate_up^T   [1024 x 2048]
  {
    GemmP p{h2b, HDIM, 0, ws_gu, HDIM, 1, 0, 1, gu, 2 * IDIM, 0,
            T_TOK, 2 * IDIM, HDIM, 1.0f, nullptr, 0, nullptr, nullptr, nullptr};
    launch_gemm(p, (2 * IDIM) / BN, T_TOK / BM, 1, stream, false, false, false, false);
  }

  // 20) shared activation
  shared_act_kernel<<<((long)T_TOK * IDIM) / 256, 256, 0, stream>>>(gu, sb);

  // 21) out = x1 + shared @ ws_down^T   (residual fused)
  {
    GemmP p{sb, IDIM, 0, ws_dn, IDIM, 1, 0, 1, out, HDIM, 0,
            T_TOK, HDIM, IDIM, 1.0f, x1, HDIM, nullptr, nullptr, nullptr};
    launch_gemm(p, HDIM / BN, T_TOK / BM, 1, stream, false, false, false, true);
  }

  // 22) out += routed experts (deterministic slot-ordered sum)
  combine_kernel<<<((long)T_TOK * HDIM) / 256, 256, 0, stream>>>(out, dmoe, rowof, topw);
}
